// UniversalFourierTransformer_1176821039334
// MI455X (gfx1250) — compile-verified, matrix operand order checked by reading, not run on hardware
//
#include <hip/hip_runtime.h>
#include <hip/hip_bf16.h>
#include <math.h>

typedef __attribute__((ext_vector_type(16))) _Float16 v16h;
typedef __attribute__((ext_vector_type(8)))  float    v8f;
typedef __attribute__((ext_vector_type(4)))  float    v4f;
typedef __attribute__((ext_vector_type(4)))  _Float16 v4h;

#define EQ_TOKEN 151

// f16 transposed-weight layout in d_ws (units: halfs). Wt[n][k] row-major.
#define OFF_MEM  0        // [64][128]
#define OFF_SUR  8192     // [128][64]
#define OFF_M2R  16384    // [128][64]
#define OFF_Q    24576    // [128][128]  n = h*32+dk, k = d
#define OFF_K    40960
#define OFF_V    57344
#define OFF_O    73728    // [128][128]  n = d_model, k = h*32+dk
#define OFF_MLP1 90112    // [512][128]
#define OFF_MLP2 155648   // [128][512]
#define OFF_UN   221184   // [160][128]  (150 padded to 160)
#define WS_HALFS 241664

__device__ __forceinline__ void lds_fence() {
  __asm__ volatile("s_wait_dscnt 0" ::: "memory");
}

__device__ __forceinline__ float fast_rcp(float x) {
  return __builtin_amdgcn_rcpf(x);   // v_rcp_f32, ~1 ulp
}

__device__ __forceinline__ float silu_f(float x) {
  return x * fast_rcp(1.0f + __expf(-x));
}

// K index of half-pair j (0..7) for lane-half h (0..1), per ISA 16-bit A/B layout:
// VGPR0..3 lanes0-15: K=0..7, lanes16-31: K=8..15; VGPR4..7: +16.
__device__ __forceinline__ int kpos(int j, int h) {
  return ((j & 4) << 2) + ((j & 3) << 1) + (h << 3);
}

// A fragment: 16xK tile (rows 0..15 of src, row-major f16, leading dim ld), K-chunk at kb.
__device__ __forceinline__ v16h frag_lds(const _Float16* src, int ld, int kb) {
  int lane = threadIdx.x & 31;
  int m = lane & 15, h = lane >> 4;
  v16h f;
#pragma unroll
  for (int j = 0; j < 8; ++j) {
    int k = kb + kpos(j, h);
    f[2 * j]     = src[m * ld + k];
    f[2 * j + 1] = src[m * ld + k + 1];
  }
  return f;
}

// B fragment from global transposed weight Wt[N][K] (row-major, f16): cols n0..n0+15, K-chunk kb.
__device__ __forceinline__ v16h frag_glb(const _Float16* W, int K, int n0, int kb) {
  int lane = threadIdx.x & 31;
  int n = n0 + (lane & 15), h = lane >> 4;
  v16h f;
#pragma unroll
  for (int j = 0; j < 8; ++j) {
    int k = kb + kpos(j, h);
    f[2 * j]     = W[n * K + k];
    f[2 * j + 1] = W[n * K + k + 1];
  }
  return f;
}

__device__ __forceinline__ v8f wmma16(v16h a, v16h b, v8f c) {
  return __builtin_amdgcn_wmma_f32_16x16x32_f16(false, a, false, b, (short)0, c,
                                                false, false);
}

// ---------------------------------------------------------------------------
// Prep: convert fp32 weights -> f16, transposed to [N][K] fragment-friendly order.
// ---------------------------------------------------------------------------
__global__ void prep_weights(const float* Wmem, const float* Wsur, const float* Wm2r,
                             const float* WQ, const float* WK, const float* WV,
                             const float* WO, const float* W1, const float* W2,
                             const float* Wun, _Float16* ws) {
  int i = blockIdx.x * blockDim.x + threadIdx.x;
  if (i >= WS_HALFS) return;
  float v;
  if (i < OFF_SUR) {                       // Wt_mem [64][128]
    int j = i - OFF_MEM, n = j >> 7, k = j & 127;
    v = Wmem[k * 64 + n];
  } else if (i < OFF_M2R) {                // Wt_sur [128][64]
    int j = i - OFF_SUR, n = j >> 6, k = j & 63;
    v = Wsur[k * 128 + n];
  } else if (i < OFF_Q) {                  // Wt_m2r [128][64]
    int j = i - OFF_M2R, n = j >> 6, k = j & 63;
    v = Wm2r[k * 128 + n];
  } else if (i < OFF_K) {                  // Wt_Q [128][128]
    int j = i - OFF_Q, n = j >> 7, k = j & 127;
    v = WQ[((n >> 5) * 128 + k) * 32 + (n & 31)];
  } else if (i < OFF_V) {
    int j = i - OFF_K, n = j >> 7, k = j & 127;
    v = WK[((n >> 5) * 128 + k) * 32 + (n & 31)];
  } else if (i < OFF_O) {
    int j = i - OFF_V, n = j >> 7, k = j & 127;
    v = WV[((n >> 5) * 128 + k) * 32 + (n & 31)];
  } else if (i < OFF_MLP1) {               // Wt_O [128][128], k = h*32+dk
    int j = i - OFF_O, n = j >> 7, k = j & 127;
    v = WO[((k >> 5) * 32 + (k & 31)) * 128 + n];
  } else if (i < OFF_MLP2) {               // Wt_mlp1 [512][128]
    int j = i - OFF_MLP1, n = j >> 7, k = j & 127;
    v = W1[k * 512 + n];
  } else if (i < OFF_UN) {                 // Wt_mlp2 [128][512]
    int j = i - OFF_MLP2, n = j >> 9, k = j & 511;
    v = W2[k * 128 + n];
  } else {                                 // Wt_un [160][128], pad n>=150 with 0
    int j = i - OFF_UN, n = j >> 7, k = j & 127;
    v = (n < 150) ? Wun[k * 150 + n] : 0.0f;
  }
  ws[i] = (_Float16)v;
}

// ---------------------------------------------------------------------------
// Fused transformer: 1 wave32 per 16 batch elements, 2 waves per block.
// Per-wave LDS (halfs): xw[64*128] | qb[64*128] | kb[64*128] | vb[64*128] | hb[16*32]
// ---------------------------------------------------------------------------
__global__ void __launch_bounds__(64)
transformer_fused(const int* pv, const int* av, const int* bv,
                  const float* tok_emb, const float* pos_emb,
                  const float* b_mem, const float* b_sur, const float* b_m2r,
                  const float* b_mlp1, const float* b_mlp2, const float* b_un,
                  const _Float16* ws, float* out) {
  extern __shared__ _Float16 smem[];
  const int lane = threadIdx.x & 31;
  const int wid  = threadIdx.x >> 5;
  const int base = (blockIdx.x * 2 + wid) * 16;

  _Float16* xw  = smem + wid * 33280;  // 64x128 residual stream (rows = s*16+m)
  _Float16* qb  = xw + 8192;           // 64x128 q (later attn-out)
  _Float16* kbf = qb + 8192;           // 64x128 k
  _Float16* vbf = kbf + 8192;          // 64x128 v
  _Float16* hb  = vbf + 8192;          // 16x32 MLP hidden bounce

  // mem-phase scratch aliases qb (used strictly before QKV)
  _Float16* Sz  = qb;                  // 16x64
  _Float16* Szu = qb + 1024;           // 16x64
  _Float16* St  = qb + 2048;           // 16x128

  const int cn = lane & 15;            // C-tile column of this lane
  const int ch = lane >> 4;            // C-tile row-half

  // ---- Phase 0: embedding + positional (vectorized float4 -> 4xf16)
  for (int row = lane; row < 64; row += 32) {
    int s = row >> 4, r = row & 15;
    int e = base + r;
    int tok = (s == 0) ? pv[e] : (s == 1) ? av[e] : (s == 2) ? bv[e] : EQ_TOKEN;
    const v4f* te4 = (const v4f*)(tok_emb + tok * 128);
    const v4f* pe4 = (const v4f*)(pos_emb + s * 128);
    v4h* dst4 = (v4h*)(xw + row * 128);
#pragma unroll 4
    for (int d = 0; d < 32; ++d) {
      v4f t = te4[d], p = pe4[d];
      v4h o;
      o[0] = (_Float16)(t[0] + p[0]);
      o[1] = (_Float16)(t[1] + p[1]);
      o[2] = (_Float16)(t[2] + p[2]);
      o[3] = (_Float16)(t[3] + p[3]);
      dst4[d] = o;
    }
  }
  lds_fence();

  // ---- Phase 1: memory path (rows = position-0 tile)
  {
    // 1a: Z = silu(x0 @ Wmem + b_mem)      [16x64]
    v16h ax0[4];
#pragma unroll
    for (int kc = 0; kc < 4; ++kc) ax0[kc] = frag_lds(xw, 128, kc * 32);
    for (int nt = 0; nt < 4; ++nt) {
      v8f c = {};
#pragma unroll
      for (int kc = 0; kc < 4; ++kc)
        c = wmma16(ax0[kc], frag_glb(ws + OFF_MEM, 128, nt * 16, kc * 32), c);
      int n = nt * 16 + cn;
      float bn = b_mem[n];
#pragma unroll
      for (int r = 0; r < 8; ++r) {
        int m = r + ch * 8;
        Sz[m * 64 + n] = (_Float16)silu_f(c[r] + bn);
      }
    }
    lds_fence();
    // 1b: T = x0 - (Z @ Wsur + b_sur)      [16x128]
    v16h az[2];
#pragma unroll
    for (int kc = 0; kc < 2; ++kc) az[kc] = frag_lds(Sz, 64, kc * 32);
    for (int nt = 0; nt < 8; ++nt) {
      v8f c = {};
#pragma unroll
      for (int kc = 0; kc < 2; ++kc)
        c = wmma16(az[kc], frag_glb(ws + OFF_SUR, 64, nt * 16, kc * 32), c);
      int n = nt * 16 + cn;
      float bn = b_sur[n];
#pragma unroll
      for (int r = 0; r < 8; ++r) {
        int m = r + ch * 8;
        St[m * 128 + n] = (_Float16)((float)xw[m * 128 + n] - (c[r] + bn));
      }
    }
    lds_fence();
    // 1c: ZU = Z + silu(T @ Wmem)          [16x64]
    v16h at[4];
#pragma unroll
    for (int kc = 0; kc < 4; ++kc) at[kc] = frag_lds(St, 128, kc * 32);
    for (int nt = 0; nt < 4; ++nt) {
      v8f c = {};
#pragma unroll
      for (int kc = 0; kc < 4; ++kc)
        c = wmma16(at[kc], frag_glb(ws + OFF_MEM, 128, nt * 16, kc * 32), c);
      int n = nt * 16 + cn;
#pragma unroll
      for (int r = 0; r < 8; ++r) {
        int m = r + ch * 8;
        Szu[m * 64 + n] = (_Float16)((float)Sz[m * 64 + n] + silu_f(c[r]));
      }
    }
    lds_fence();
    // 1d: MEM = ZU @ Wm2r + b_m2r ; x[s,:,:] += MEM for all 4 positions
    v16h azu[2];
#pragma unroll
    for (int kc = 0; kc < 2; ++kc) azu[kc] = frag_lds(Szu, 64, kc * 32);
    for (int nt = 0; nt < 8; ++nt) {
      v8f c = {};
#pragma unroll
      for (int kc = 0; kc < 2; ++kc)
        c = wmma16(azu[kc], frag_glb(ws + OFF_M2R, 64, nt * 16, kc * 32), c);
      int n = nt * 16 + cn;
      float bn = b_m2r[n];
#pragma unroll
      for (int r = 0; r < 8; ++r) {
        int m = r + ch * 8;
        float mv = c[r] + bn;
#pragma unroll
        for (int s = 0; s < 4; ++s) {
          int idx = (s * 16 + m) * 128 + n;
          xw[idx] = (_Float16)((float)xw[idx] + mv);
        }
      }
    }
  }
  lds_fence();

  // ---- Phase 2: QKV projections (q pre-scaled by 1/sqrt(32))
  for (int st = 0; st < 4; ++st) {
    v16h ax[4];
#pragma unroll
    for (int kc = 0; kc < 4; ++kc) ax[kc] = frag_lds(xw + st * 16 * 128, 128, kc * 32);
#pragma unroll
    for (int w = 0; w < 3; ++w) {
      const _Float16* Wt = (w == 0) ? ws + OFF_Q : (w == 1) ? ws + OFF_K : ws + OFF_V;
      _Float16* dst = (w == 0) ? qb : (w == 1) ? kbf : vbf;
      float scale = (w == 0) ? 0.17677669529663687f : 1.0f;
      for (int nt = 0; nt < 8; ++nt) {
        v8f c = {};
#pragma unroll
        for (int kc = 0; kc < 4; ++kc)
          c = wmma16(ax[kc], frag_glb(Wt, 128, nt * 16, kc * 32), c);
        int n = nt * 16 + cn;
#pragma unroll
        for (int r = 0; r < 8; ++r) {
          int m = r + ch * 8;
          dst[(st * 16 + m) * 128 + n] = (_Float16)(c[r] * scale);
        }
      }
    }
  }
  lds_fence();

  // ---- Phase 3: 4x4 softmax attention per (element, head), VALU.
  // lane -> element e = lane>>1, handles 2 heads; writes attn-out back into qb.
  {
    int e = lane >> 1;
    int h0 = (lane & 1) * 2;
#pragma unroll
    for (int hh = 0; hh < 2; ++hh) {
      int cb = (h0 + hh) * 32;
      float sc[4][4];
#pragma unroll
      for (int s = 0; s < 4; ++s)
#pragma unroll
        for (int t = 0; t < 4; ++t) {
          float d = 0.0f;
          for (int k2 = 0; k2 < 32; ++k2)
            d += (float)qb[(s * 16 + e) * 128 + cb + k2] *
                 (float)kbf[(t * 16 + e) * 128 + cb + k2];
          sc[s][t] = d;
        }
      float p[4][4];
#pragma unroll
      for (int s = 0; s < 4; ++s) {
        float mx = fmaxf(fmaxf(sc[s][0], sc[s][1]), fmaxf(sc[s][2], sc[s][3]));
        float sum = 0.0f;
#pragma unroll
        for (int t = 0; t < 4; ++t) { p[s][t] = __expf(sc[s][t] - mx); sum += p[s][t]; }
        float inv = fast_rcp(sum);
#pragma unroll
        for (int t = 0; t < 4; ++t) p[s][t] *= inv;
      }
#pragma unroll
      for (int s = 0; s < 4; ++s)
        for (int k2 = 0; k2 < 32; ++k2) {
          float a0 = 0.0f;
#pragma unroll
          for (int t = 0; t < 4; ++t)
            a0 += p[s][t] * (float)vbf[(t * 16 + e) * 128 + cb + k2];
          qb[(s * 16 + e) * 128 + cb + k2] = (_Float16)a0;
        }
    }
  }
  lds_fence();

  // ---- Phase 4: x += attn @ W_O
  for (int st = 0; st < 4; ++st) {
    v16h aa[4];
#pragma unroll
    for (int kc = 0; kc < 4; ++kc) aa[kc] = frag_lds(qb + st * 16 * 128, 128, kc * 32);
    for (int nt = 0; nt < 8; ++nt) {
      v8f c = {};
#pragma unroll
      for (int kc = 0; kc < 4; ++kc)
        c = wmma16(aa[kc], frag_glb(ws + OFF_O, 128, nt * 16, kc * 32), c);
      int n = nt * 16 + cn;
#pragma unroll
      for (int r = 0; r < 8; ++r) {
        int m = r + ch * 8;
        int idx = (st * 16 + m) * 128 + n;
        xw[idx] = (_Float16)((float)xw[idx] + c[r]);
      }
    }
  }
  lds_fence();

  // ---- Phase 5: x += relu(x @ W1 + b1) @ W2 + b2 (hidden streamed 32 at a time)
  for (int st = 0; st < 4; ++st) {
    v16h ax[4];
#pragma unroll
    for (int kc = 0; kc < 4; ++kc) ax[kc] = frag_lds(xw + st * 16 * 128, 128, kc * 32);
    v8f zero = {};
    v8f acc[8];
#pragma unroll
    for (int nt = 0; nt < 8; ++nt) acc[nt] = zero;
    for (int hc = 0; hc < 16; ++hc) {
#pragma unroll
      for (int hf = 0; hf < 2; ++hf) {
        v8f c = {};
#pragma unroll
        for (int kc = 0; kc < 4; ++kc)
          c = wmma16(ax[kc], frag_glb(ws + OFF_MLP1, 128, hc * 32 + hf * 16, kc * 32), c);
        int n = hc * 32 + hf * 16 + cn;
        float bn = b_mlp1[n];
#pragma unroll
        for (int r = 0; r < 8; ++r) {
          int m = r + ch * 8;
          hb[m * 32 + hf * 16 + cn] = (_Float16)fmaxf(c[r] + bn, 0.0f);
        }
      }
      lds_fence();
      v16h ah = frag_lds(hb, 32, 0);
      lds_fence();
#pragma unroll
      for (int nt = 0; nt < 8; ++nt)
        acc[nt] = wmma16(ah, frag_glb(ws + OFF_MLP2, 512, nt * 16, hc * 32), acc[nt]);
    }
#pragma unroll
    for (int nt = 0; nt < 8; ++nt) {
      int n = nt * 16 + cn;
      float bn = b_mlp2[n];
#pragma unroll
      for (int r = 0; r < 8; ++r) {
        int m = r + ch * 8;
        int idx = (st * 16 + m) * 128 + n;
        xw[idx] = (_Float16)((float)xw[idx] + acc[nt][r] + bn);
      }
    }
  }
  lds_fence();

  // ---- Phase 6: logits = x[:,3,:] @ W_un + b_un  (N padded 150->160)
  {
    v16h ax[4];
#pragma unroll
    for (int kc = 0; kc < 4; ++kc) ax[kc] = frag_lds(xw + 3 * 16 * 128, 128, kc * 32);
    for (int nt = 0; nt < 10; ++nt) {
      v8f c = {};
#pragma unroll
      for (int kc = 0; kc < 4; ++kc)
        c = wmma16(ax[kc], frag_glb(ws + OFF_UN, 128, nt * 16, kc * 32), c);
      int n = nt * 16 + cn;
      if (n < 150) {
        float bn = b_un[n];
#pragma unroll
        for (int r = 0; r < 8; ++r) {
          int m = r + ch * 8;
          out[(size_t)(base + m) * 150 + n] = c[r] + bn;
        }
      }
    }
  }
}

// ---------------------------------------------------------------------------
extern "C" void kernel_launch(void* const* d_in, const int* in_sizes, int n_in,
                              void* d_out, int out_size, void* d_ws, size_t ws_size,
                              hipStream_t stream) {
  const int*   pv   = (const int*)d_in[0];
  const int*   av   = (const int*)d_in[1];
  const int*   bv   = (const int*)d_in[2];
  const float* tok  = (const float*)d_in[3];
  const float* pos  = (const float*)d_in[4];
  const float* Wmem = (const float*)d_in[5];
  const float* bmem = (const float*)d_in[6];
  const float* Wsur = (const float*)d_in[7];
  const float* bsur = (const float*)d_in[8];
  const float* Wm2r = (const float*)d_in[9];
  const float* bm2r = (const float*)d_in[10];
  const float* WQ   = (const float*)d_in[11];
  const float* WK   = (const float*)d_in[12];
  const float* WV   = (const float*)d_in[13];
  const float* WO   = (const float*)d_in[14];
  const float* W1   = (const float*)d_in[15];
  const float* b1   = (const float*)d_in[16];
  const float* W2   = (const float*)d_in[17];
  const float* b2   = (const float*)d_in[18];
  const float* Wun  = (const float*)d_in[19];
  const float* bun  = (const float*)d_in[20];
  _Float16* ws = (_Float16*)d_ws;
  float* out = (float*)d_out;
  const int B = in_sizes[0];

  prep_weights<<<(WS_HALFS + 255) / 256, 256, 0, stream>>>(
      Wmem, Wsur, Wm2r, WQ, WK, WV, WO, W1, W2, Wun, ws);

  int blocks = B / 32;                         // 2 waves/block x 16 elems/wave
  size_t shmem = (size_t)2 * 33280 * sizeof(_Float16);  // ~130 KB of 320 KB WGP LDS
  transformer_fused<<<blocks, 64, shmem, stream>>>(
      pv, av, bv, tok, pos, bmem, bsur, bm2r, b1, b2, bun, ws, out);
}
